// SingleGNNModel_44504451121837
// MI455X (gfx1250) — matrix operand order repaired and checked by reading, not compiled, not run on hardware
//
#include <hip/hip_runtime.h>

typedef __attribute__((ext_vector_type(16))) _Float16 v16h;
typedef __attribute__((ext_vector_type(8)))  float    v8f;

#define DH 128

// ---------------------------------------------------------------- utilities
__global__ void zero4_kernel(float4* __restrict__ p, long n4) {
  long i = (long)blockIdx.x * blockDim.x + threadIdx.x;
  long stride = (long)gridDim.x * blockDim.x;
  float4 z = {0.f, 0.f, 0.f, 0.f};
  for (; i < n4; i += stride) p[i] = z;
}

// ---------------------------------------------------------------- degrees
__global__ void degree_kernel(const int* __restrict__ src, const int* __restrict__ dst,
                              float* __restrict__ degO, float* __restrict__ degI, int E) {
  int e = blockIdx.x * blockDim.x + threadIdx.x;
  if (e < E) {
    atomicAdd(&degO[src[e]], 1.0f);
    atomicAdd(&degI[dst[e]], 1.0f);
  }
}

__global__ void rsqrt_kernel(float* __restrict__ d, int n) {
  int i = blockIdx.x * blockDim.x + threadIdx.x;
  if (i < n) d[i] = rsqrtf(fmaxf(d[i], 1.0f));
}

// ------------------------------------------------- edge scatter (wave/edge)
// m[dst, :] += h[src, :] * inv_src[src]   (128 floats/edge, lane does float4)
__global__ void scatter_kernel(const float* __restrict__ h, const int* __restrict__ src,
                               const int* __restrict__ dst, const float* __restrict__ invs,
                               float* __restrict__ m, int E) {
  int lane = threadIdx.x & 31;
  int e = blockIdx.x * (blockDim.x >> 5) + (threadIdx.x >> 5);
  if (e >= E) return;
  int s = src[e], d = dst[e];
  float sc = invs[s];
  const float4 v = *(const float4*)(h + (long)s * DH + lane * 4);
  float* o = m + (long)d * DH + lane * 4;
  atomicAdd(o + 0, v.x * sc);
  atomicAdd(o + 1, v.y * sc);
  atomicAdd(o + 2, v.z * sc);
  atomicAdd(o + 3, v.w * sc);
}

// ------------------------------------- weight fp32 -> f16 in B-frag order
// B operand order used by this code (16-bit 32x16, wave32, contiguous k): element h of lane l:
//   k = kc*32 + (l&16) + h , n = ct*16 + (l&15)
// stored as out[((ct*kcMax + kc)*32 + l)*16 + h]
__global__ void convert_w_kernel(const float* __restrict__ W, _Float16* __restrict__ out,
                                 int K, int Nn) {
  int idx = blockIdx.x * blockDim.x + threadIdx.x;
  if (idx >= K * Nn) return;
  int kcMax = K >> 5;
  int h = idx & 15;
  int l = (idx >> 4) & 31;
  int rest = idx >> 9;
  int kc = rest % kcMax;
  int ct = rest / kcMax;
  int k = kc * 32 + (l & 16) + h;
  int n = ct * 16 + (l & 15);
  out[idx] = (_Float16)W[k * Nn + n];
}

// --------------------------------------------------------- WMMA GEMM+ReLU
// Y[M,Nout] = act( (X[M,K] * scale[row]?) @ W + bias )
// block = 256 threads (8 waves); blockIdx.x = 16-row tile; blockIdx.y*128 cols.
// A staged in LDS in this code's contiguous-k 16x32 A-operand order (same k order as B):
//   row r, col c: kc=c/32, cl=c%32 ->
//     cl<16: l=r,    h=cl       ; else    : l=r+16, h=cl-16
__global__ void gemm_wmma_kernel(const float* __restrict__ X, const float* __restrict__ scale,
                                 const _Float16* __restrict__ wfrag,
                                 const float* __restrict__ bias, float* __restrict__ Y,
                                 int K, int Nout, int relu) {
  __shared__ _Float16 sA[8192];  // up to K=512: 16 rows, frag-ordered
  const int kcMax = K >> 5;
  const int tid = threadIdx.x;
  const int lane = tid & 31;
  const int wv = tid >> 5;
  const long rowBase = (long)blockIdx.x * 16;

  for (int e = tid; e < 16 * K; e += 256) {
    int r = e / K;
    int c = e - r * K;
    long row = rowBase + r;
    float v = X[row * K + c];
    if (scale) v *= scale[row];
    int kc = c >> 5;
    int cl = c & 31;
    int l, h;
    if (cl < 16)      { l = r;      h = cl;      }
    else              { l = r + 16; h = cl - 16; }
    sA[(kc * 32 + l) * 16 + h] = (_Float16)v;
  }
  __syncthreads();

  const int ct = blockIdx.y * 8 + wv;  // global 16-col tile
  v8f acc = {};
  const _Float16* wp = wfrag + ((long)(ct * kcMax) * 32 + lane) * 16;
  for (int kc = 0; kc < kcMax; ++kc) {
    v16h a = *(const v16h*)&sA[(kc * 32 + lane) * 16];
    v16h b = *(const v16h*)&wp[kc * 512];
    acc = __builtin_amdgcn_wmma_f32_16x16x32_f16(false, a, false, b, (short)0, acc,
                                                 false, false);
  }

  const int n = ct * 16 + (lane & 15);
  const float bb = bias[n];
  const int mofs = (lane < 16) ? 0 : 8;
#pragma unroll
  for (int g = 0; g < 8; ++g) {
    float v = acc[g] + bb;
    if (relu) v = fmaxf(v, 0.0f);
    Y[(rowBase + g + mofs) * Nout + n] = v;
  }
}

// ----------------------------------------------------------------- pooling
__global__ void pool_scatter_kernel(const float* __restrict__ h, const int* __restrict__ gid,
                                    float* __restrict__ emb, float* __restrict__ counts, int N) {
  int lane = threadIdx.x & 31;
  int i = blockIdx.x * (blockDim.x >> 5) + (threadIdx.x >> 5);
  if (i >= N) return;
  int g = gid[i];
  float4 v = *(const float4*)(h + (long)i * DH + lane * 4);
  float* o = emb + (long)g * DH + lane * 4;
  atomicAdd(o + 0, v.x);
  atomicAdd(o + 1, v.y);
  atomicAdd(o + 2, v.z);
  atomicAdd(o + 3, v.w);
  if (lane == 0) atomicAdd(&counts[g], 1.0f);
}

__global__ void pool_div_kernel(float* __restrict__ emb, const float* __restrict__ counts,
                                int total) {
  int i = blockIdx.x * blockDim.x + threadIdx.x;
  if (i < total) emb[i] /= fmaxf(counts[i / DH], 1.0f);
}

// ------------------------------------------------------ final [K]->1 head
__global__ void final_kernel(const float* __restrict__ x, const float* __restrict__ w,
                             const float* __restrict__ b, float* __restrict__ out,
                             int G, int K) {
  int g = blockIdx.x * blockDim.x + threadIdx.x;
  if (g >= G) return;
  float s = b[0];
  for (int k = 0; k < K; ++k) s += x[(long)g * K + k] * w[k];
  out[g] = s;
}

// ----------------------------------------------------------------- launch
extern "C" void kernel_launch(void* const* d_in, const int* in_sizes, int n_in,
                              void* d_out, int out_size, void* d_ws, size_t ws_size,
                              hipStream_t stream) {
  const int N = 100000, E = 1600000, G = 2048, MLPH = 512;

  const float* feats = (const float*)d_in[0];
  const int*   src   = (const int*)d_in[1];
  const int*   dst   = (const int*)d_in[2];
  const int*   gid   = (const int*)d_in[3];
  const float* W0 = (const float*)d_in[4];  const float* b0 = (const float*)d_in[5];
  const float* W1 = (const float*)d_in[6];  const float* b1 = (const float*)d_in[7];
  const float* W2 = (const float*)d_in[8];  const float* b2 = (const float*)d_in[9];
  const float* Wm0 = (const float*)d_in[10]; const float* bm0 = (const float*)d_in[11];
  const float* Wm1 = (const float*)d_in[12]; const float* bm1 = (const float*)d_in[13];
  const float* Wm2 = (const float*)d_in[14]; const float* bm2 = (const float*)d_in[15];

  char* ws = (char*)d_ws;
  auto alloc = [&](size_t bytes) -> char* {
    char* p = ws;
    ws += (bytes + 255) & ~(size_t)255;
    return p;
  };
  float* bufB = (float*)alloc((size_t)N * DH * 4);   // scatter target m
  float* bufA = (float*)alloc((size_t)N * DH * 4);   // layer activations
  float* invO = (float*)alloc((size_t)N * 4);        // deg_out -> rsqrt (src norm)
  float* invI = (float*)alloc((size_t)N * 4);        // deg_in  -> rsqrt (dst norm)
  float* emb  = (float*)alloc((size_t)G * DH * 4);
  float* cnts = (float*)alloc((size_t)G * 4);
  float* x1   = (float*)alloc((size_t)G * MLPH * 4);
  float* x2   = (float*)alloc((size_t)G * (MLPH / 2) * 4);
  _Float16* f0  = (_Float16*)alloc((size_t)DH * DH * 2);
  _Float16* f1  = (_Float16*)alloc((size_t)DH * DH * 2);
  _Float16* f2  = (_Float16*)alloc((size_t)DH * DH * 2);
  _Float16* fm0 = (_Float16*)alloc((size_t)DH * MLPH * 2);
  _Float16* fm1 = (_Float16*)alloc((size_t)MLPH * (MLPH / 2) * 2);

  // degrees -> rsqrt norms
  zero4_kernel<<<256, 256, 0, stream>>>((float4*)invO, N / 4);
  zero4_kernel<<<256, 256, 0, stream>>>((float4*)invI, N / 4);
  degree_kernel<<<(E + 255) / 256, 256, 0, stream>>>(src, dst, invO, invI, E);
  rsqrt_kernel<<<(N + 255) / 256, 256, 0, stream>>>(invO, N);
  rsqrt_kernel<<<(N + 255) / 256, 256, 0, stream>>>(invI, N);

  // weight fragment pre-conversion (fp32 -> f16, WMMA B-fragment order)
  convert_w_kernel<<<(DH * DH) / 256, 256, 0, stream>>>(W0, f0, DH, DH);
  convert_w_kernel<<<(DH * DH) / 256, 256, 0, stream>>>(W1, f1, DH, DH);
  convert_w_kernel<<<(DH * DH) / 256, 256, 0, stream>>>(W2, f2, DH, DH);
  convert_w_kernel<<<(DH * MLPH) / 256, 256, 0, stream>>>(Wm0, fm0, DH, MLPH);
  convert_w_kernel<<<(MLPH * (MLPH / 2)) / 256, 256, 0, stream>>>(Wm1, fm1, MLPH, MLPH / 2);

  // 3 GraphConv layers: zero m, scatter(h*inv_src), gemm(m*inv_dst @ W + b, relu)
  const float*    layerIn[3] = { feats, bufA, bufA };
  const _Float16* layerW[3]  = { f0, f1, f2 };
  const float*    layerB[3]  = { b0, b1, b2 };
  for (int layer = 0; layer < 3; ++layer) {
    zero4_kernel<<<2048, 256, 0, stream>>>((float4*)bufB, (long)N * DH / 4);
    scatter_kernel<<<(E + 7) / 8, 256, 0, stream>>>(layerIn[layer], src, dst, invO, bufB, E);
    gemm_wmma_kernel<<<dim3(N / 16, 1), 256, 0, stream>>>(bufB, invI, layerW[layer],
                                                          layerB[layer], bufA, DH, DH, 1);
  }

  // mean pool per graph
  zero4_kernel<<<256, 256, 0, stream>>>((float4*)emb, (long)G * DH / 4);
  zero4_kernel<<<8, 256, 0, stream>>>((float4*)cnts, G / 4);
  pool_scatter_kernel<<<(N + 7) / 8, 256, 0, stream>>>(bufA, gid, emb, cnts, N);
  pool_div_kernel<<<(G * DH) / 256, 256, 0, stream>>>(emb, cnts, G * DH);

  // MLP head
  gemm_wmma_kernel<<<dim3(G / 16, MLPH / 128), 256, 0, stream>>>(emb, nullptr, fm0, bm0,
                                                                 x1, DH, MLPH, 1);
  gemm_wmma_kernel<<<dim3(G / 16, (MLPH / 2) / 128), 256, 0, stream>>>(x1, nullptr, fm1, bm1,
                                                                       x2, MLPH, MLPH / 2, 1);
  final_kernel<<<(G + 255) / 256, 256, 0, stream>>>(x2, Wm2, bm2, (float*)d_out, G, MLPH / 2);
}
